// Attention_3d_32280974197398
// MI455X (gfx1250) — matrix operand order repaired and checked by reading, not compiled, not run on hardware
//
#include <hip/hip_runtime.h>

// ---------------- problem constants ----------------
constexpr int kC  = 64;
constexpr int kD  = 48;
constexpr int kW  = 128;
constexpr int kH  = 128;
constexpr int kWH = kW * kH;               // 16384
constexpr int kNV = kD * kW * kH;          // 786432 voxels per channel
constexpr float kQScale = 0.5946035575013605f;  // 8^(-1/4)

// ---------------- vector types ----------------
typedef __attribute__((ext_vector_type(16))) __bf16        v16bf;
typedef __attribute__((ext_vector_type(8)))  float         v8f;
typedef __attribute__((ext_vector_type(4)))  float         v4f;
typedef __attribute__((ext_vector_type(4)))  unsigned int  v4u;
typedef __attribute__((ext_vector_type(4)))  unsigned int  u32x4;
typedef __attribute__((ext_vector_type(8)))  int           i32x8;
typedef __attribute__((ext_vector_type(4)))  int           i32x4;

union BF8 { v4u v; unsigned short s[8]; };
union V16 { v4u u[2]; v16bf v; };

// Tensor Data Mover availability (this toolchain: clang-23, 6-arg builtin)
#if defined(__has_builtin)
#if __has_builtin(__builtin_amdgcn_tensor_load_to_lds) && \
    __has_builtin(__builtin_amdgcn_s_wait_tensorcnt)
#define USE_TDM 1
#endif
#endif
#ifndef USE_TDM
#define USE_TDM 0
#endif

__device__ inline unsigned short f2bf(float f) {
  unsigned u = __float_as_uint(f);
  u += 0x7fffu + ((u >> 16) & 1u);           // round-to-nearest-even
  return (unsigned short)(u >> 16);
}

__device__ inline v8f zero8() {
  v8f z;
#pragma unroll
  for (int i = 0; i < 8; ++i) z[i] = 0.f;
  return z;
}

// A fragment 16x32 bf16 from LDS row-major [m][k] (stride in elements, %8==0).
// ISA layout: lanes 0-15 -> K 0..7 & 16..23 ; lanes 16-31 -> K 8..15 & 24..31.
template <int STRIDE>
__device__ inline v16bf load_matA(const unsigned short* smem, int mb, int kb) {
  const int l  = threadIdx.x & 31;
  const int m  = mb + (l & 15);
  const int ko = kb + ((l >> 4) << 3);
  const unsigned short* p = smem + m * STRIDE + ko;
  V16 r;
  r.u[0] = *(const v4u*)(p);         // K = ko .. ko+7
  r.u[1] = *(const v4u*)(p + 16);    // K = ko+16 .. ko+23
  return r.v;
}

// B fragment 32x16 bf16 from LDS stored transposed [n][k] (stride %8==0).
// ISA layout: lanes 0-15 -> K 0..7 & 16..23 ; lanes 16-31 -> K 8..15 & 24..31.
template <int STRIDE>
__device__ inline v16bf load_matB(const unsigned short* smem, int kb, int nb) {
  const int l  = threadIdx.x & 31;
  const int n  = nb + (l & 15);
  const int ko = kb + ((l >> 4) << 3);
  const unsigned short* p = smem + n * STRIDE + ko;
  V16 r;
  r.u[0] = *(const v4u*)(p);         // K = ko .. ko+7
  r.u[1] = *(const v4u*)(p + 16);    // K = ko+16 .. ko+23
  return r.v;
}

__device__ inline v8f wmma_bf16(v16bf a, v16bf b, v8f c) {
  return __builtin_amdgcn_wmma_f32_16x16x32_bf16(
      false, a, false, b, (short)0, c, false, false);
}

#if USE_TDM
// TDM: load a contiguous 128x128 2-byte 2-D tile into LDS at lds_off,
// inserting 16B of LDS padding per 256B row -> LDS row stride 272B (136 elems).
__device__ inline void tdm_load_tile128x128_bf16(unsigned lds_off, const void* gptr) {
  const unsigned long long ga = (unsigned long long)(uintptr_t)gptr;
  u32x4 g0;
  g0[0] = 1u;                                     // count=1, user descriptor
  g0[1] = lds_off;                                // LDS byte address
  g0[2] = (unsigned)(ga & 0xffffffffu);           // global_addr[31:0]
  g0[3] = (unsigned)((ga >> 32) & 0x1ffffffu)     // global_addr[56:32]
        | 0x80000000u;                            // type=2 ("image")
  i32x8 g1;
  g1[0] = (1 << 16)                               // data_size = 2 bytes
        | (1 << 20)                               // pad_enable
        | (5 << 22)                               // pad_interval: every 256B
        | (3 << 25);                              // pad_amount: 4 DWORDs (16B)
  g1[1] = (kW & 0xffff) << 16;                    // tensor_dim0[15:0]   (=128)
  g1[2] = (kW & 0xffff) << 16;                    // tensor_dim1[15:0]   (=128)
  g1[3] = (kW & 0xffff) << 16;                    // tile_dim0           (=128)
  g1[4] = kW;                                     // tile_dim1 (=128), tile_dim2=0
  g1[5] = kW;                                     // tensor_dim0_stride[31:0]
  g1[6] = 0;
  g1[7] = 0;
  i32x4 z4;
  z4[0] = z4[1] = z4[2] = z4[3] = 0;              // 2-D: groups 2/3 unused
  i32x8 z8;
#pragma unroll
  for (int i = 0; i < 8; ++i) z8[i] = 0;
  // clang-23 / therock-10.0 signature: (g0, g1, g2, g3, g4, cpol)
  __builtin_amdgcn_tensor_load_to_lds(g0, g1, z4, z4, z8, 0);
}
#endif

// ============================================================
// Kernel 1: fused QKV projection.  out[192,128-tile] = Wqkv @ x-tile
// one workgroup (8 waves) per 128 voxels.
// ============================================================
__global__ __launch_bounds__(256) void qkv_proj_kernel(
    const float* __restrict__ x,
    const float* __restrict__ wq, const float* __restrict__ bq,
    const float* __restrict__ wk, const float* __restrict__ bk,
    const float* __restrict__ wv, const float* __restrict__ bv,
    unsigned short* __restrict__ qbuf,
    unsigned short* __restrict__ kbuf,
    unsigned short* __restrict__ vbuf)
{
  __shared__ alignas(16) unsigned short wsm[192 * 72];  // stacked Wq|Wk|Wv, [m][k]
  __shared__ alignas(16) unsigned short xsm[128 * 72];  // x tile, [n][k]
  __shared__ float bsm[192];

  const int tid = threadIdx.x;
  const int p0  = blockIdx.x * 128;

  // weights -> LDS (bf16)
#pragma unroll
  for (int pass = 0; pass < 12; ++pass) {
    int i = (pass * 256 + tid) * 4;
    int row = i >> 6, col = i & 63;
    const float* src = (row < 64) ? (wq + row * 64)
                     : (row < 128) ? (wk + (row - 64) * 64)
                                   : (wv + (row - 128) * 64);
    v4f w4 = *(const v4f*)(src + col);
#pragma unroll
    for (int j = 0; j < 4; ++j) wsm[row * 72 + col + j] = f2bf(w4[j]);
  }
  if (tid < 192)
    bsm[tid] = (tid < 64) ? bq[tid] : (tid < 128) ? bk[tid - 64] : bv[tid - 128];

  // x tile -> LDS transposed [voxel][channel]
#pragma unroll
  for (int pass = 0; pass < 8; ++pass) {
    int i = pass * 1024 + tid * 4;
    int cch = i >> 7, col = i & 127;
    v4f xv = *(const v4f*)(x + (size_t)cch * kNV + p0 + col);
#pragma unroll
    for (int j = 0; j < 4; ++j) xsm[(col + j) * 72 + cch] = f2bf(xv[j]);
  }
  __syncthreads();

  const int wvid = tid >> 5;          // wave id 0..7 -> 16-column strip
  const int lane = tid & 31;
  const int nn   = wvid * 16 + (lane & 15);
  const int mhi  = (lane >> 4) << 3;

  for (int mb = 0; mb < 12; ++mb) {
    v8f acc = zero8();
#pragma unroll
    for (int kk = 0; kk < 64; kk += 32) {
      v16bf a = load_matA<72>(wsm, mb * 16, kk);
      v16bf b = load_matB<72>(xsm, kk, wvid * 16);
      acc = wmma_bf16(a, b, acc);
    }
    // Tile rows are 16-aligned and the q/k/v boundary is 64-aligned, so the
    // destination matrix and scale are UNIFORM per tile (no lane selects).
    const int mat = mb >> 2;                       // 0=q, 1=k, 2=v
    unsigned short* dstb = (mat == 0) ? qbuf : (mat == 1) ? kbuf : vbuf;
    const float scale = (mat == 0) ? kQScale : 1.f;
    unsigned short* dst = dstb + (size_t)((mb & 3) * 16) * kNV + p0 + nn;
#pragma unroll
    for (int r = 0; r < 8; ++r) {
      int gm = mb * 16 + r + mhi;                  // 0..191 (bias index)
      float val = (acc[r] + bsm[gm]) * scale;      // scale applied after bias
      dst[(size_t)(r + mhi) * kNV] = f2bf(val);
    }
  }
}

// ============================================================
// Kernel 2: fused attention per (channel, depth) slice.
// S = q k^T ; softmax rows ; O = P v.  One workgroup per slice.
// Wave i owns rows 16i..16i+15 of S/P/O.
// q/k/v tiles staged into LDS by the Tensor Data Mover when available.
// ============================================================
__global__ __launch_bounds__(256) void attn_kernel(
    const unsigned short* __restrict__ qbuf,
    const unsigned short* __restrict__ kbuf,
    const unsigned short* __restrict__ vbuf,
    unsigned short* __restrict__ obuf)
{
  extern __shared__ unsigned short smem[];
  unsigned short* qsm = smem;                 // [w][h]   stride 136
  unsigned short* ksm = qsm + 128 * 136;      // [w'][h]  stride 136
  unsigned short* vsm = ksm + 128 * 136;      // [h][w']  (transposed) stride 136
  unsigned short* psm = vsm + 128 * 136;      // [w][w']  stride 136 (also v staging)

  const int tid = threadIdx.x;
  const int d = blockIdx.x;
  const int c = blockIdx.y;
  const size_t sb = (size_t)c * kNV + (size_t)d * kWH;
  const unsigned short* qs = qbuf + sb;
  const unsigned short* ks = kbuf + sb;
  const unsigned short* vs = vbuf + sb;

#if USE_TDM
  // One wave issues three tensor DMAs: q, k row-major into qsm/ksm, and v
  // row-major into the (still unused) psm region for an LDS-side transpose.
  if (tid < 32) {
    tdm_load_tile128x128_bf16(0u,          qs);
    tdm_load_tile128x128_bf16(34816u,      ks);     // 128*136*2
    tdm_load_tile128x128_bf16(104448u,     vs);     // 3*128*136*2 (psm)
    __builtin_amdgcn_s_wait_tensorcnt(0);
  }
  __syncthreads();                                  // publish TDM-written LDS
  // transpose v: psm[w'][h] -> vsm[h][w']  (LDS -> regs -> LDS)
#pragma unroll
  for (int pass = 0; pass < 8; ++pass) {
    int i = pass * 2048 + tid * 8;
    int w = i >> 7, h = i & 127;
    BF8 vv; vv.v = *(const v4u*)(&psm[w * 136 + h]);
#pragma unroll
    for (int j = 0; j < 8; ++j) vsm[(h + j) * 136 + w] = vv.s[j];
  }
  __syncthreads();   // transpose reads of psm done before P overwrites it
#else
#pragma unroll
  for (int pass = 0; pass < 8; ++pass) {
    int i = pass * 2048 + tid * 8;
    int w = i >> 7, h = i & 127;
    *(v4u*)(&qsm[w * 136 + h]) = *(const v4u*)(qs + i);
    *(v4u*)(&ksm[w * 136 + h]) = *(const v4u*)(ks + i);
    BF8 vv; vv.v = *(const v4u*)(vs + i);
#pragma unroll
    for (int j = 0; j < 8; ++j) vsm[(h + j) * 136 + w] = vv.s[j];
  }
  __syncthreads();
#endif

  const int wvid = tid >> 5;
  const int lane = tid & 31;
  const int mb   = wvid * 16;                // this wave's row block
  const int mhi  = (lane >> 4) << 3;

  // ---- S = q @ k^T : 8 tiles x K=128 ----
  v8f accS[8];
#pragma unroll
  for (int t = 0; t < 8; ++t) accS[t] = zero8();
#pragma unroll
  for (int kk = 0; kk < 128; kk += 32) {
    v16bf a = load_matA<136>(qsm, mb, kk);
#pragma unroll
    for (int t = 0; t < 8; ++t) {
      v16bf b = load_matB<136>(ksm, kk, t * 16);
      accS[t] = wmma_bf16(a, b, accS[t]);
    }
  }

  // ---- row softmax in registers.  For acc VGPR r: lanes 0-15 hold row r,
  // lanes 16-31 hold row r+8 -> xor-masks 1,2,4,8 reduce each half-wave. ----
#pragma unroll
  for (int r = 0; r < 8; ++r) {
    float mx = -3.4028235e38f;
#pragma unroll
    for (int t = 0; t < 8; ++t) mx = fmaxf(mx, accS[t][r]);
#pragma unroll
    for (int off = 1; off < 16; off <<= 1) mx = fmaxf(mx, __shfl_xor(mx, off, 32));
    float sum = 0.f;
#pragma unroll
    for (int t = 0; t < 8; ++t) {
      float e = __expf(accS[t][r] - mx);
      accS[t][r] = e;
      sum += e;
    }
#pragma unroll
    for (int off = 1; off < 16; off <<= 1) sum += __shfl_xor(sum, off, 32);
    float inv = 1.f / sum;
#pragma unroll
    for (int t = 0; t < 8; ++t) accS[t][r] *= inv;
  }

  // ---- P -> LDS (bf16) ----
#pragma unroll
  for (int t = 0; t < 8; ++t)
#pragma unroll
    for (int r = 0; r < 8; ++r)
      psm[(mb + r + mhi) * 136 + t * 16 + (lane & 15)] = f2bf(accS[t][r]);
  __syncthreads();   // P is re-read across lanes; each wave reads only its own 16 rows

  // ---- O = P @ v ----
  v8f accO[8];
#pragma unroll
  for (int t = 0; t < 8; ++t) accO[t] = zero8();
#pragma unroll
  for (int kk = 0; kk < 128; kk += 32) {
    v16bf a = load_matA<136>(psm, mb, kk);
#pragma unroll
    for (int t = 0; t < 8; ++t) {
      v16bf b = load_matB<136>(vsm, kk, t * 16);
      accO[t] = wmma_bf16(a, b, accO[t]);
    }
  }

  unsigned short* os = obuf + sb;     // overwrite q slice (already consumed)
#pragma unroll
  for (int t = 0; t < 8; ++t)
#pragma unroll
    for (int r = 0; r < 8; ++r)
      os[(mb + r + mhi) * 128 + t * 16 + (lane & 15)] = f2bf(accO[t][r]);
}

// ============================================================
// Kernel 3: output projection.  out = Wo @ o + bo  (fp32 out)
// ============================================================
__global__ __launch_bounds__(256) void out_proj_kernel(
    const unsigned short* __restrict__ obuf,
    const float* __restrict__ wo, const float* __restrict__ bo,
    float* __restrict__ out)
{
  __shared__ alignas(16) unsigned short wsm[64 * 72];
  __shared__ alignas(16) unsigned short xsm[128 * 72];
  __shared__ float bsm[64];

  const int tid = threadIdx.x;
  const int p0  = blockIdx.x * 128;

#pragma unroll
  for (int pass = 0; pass < 4; ++pass) {
    int i = (pass * 256 + tid) * 4;
    int row = i >> 6, col = i & 63;
    v4f w4 = *(const v4f*)(wo + row * 64 + col);
#pragma unroll
    for (int j = 0; j < 4; ++j) wsm[row * 72 + col + j] = f2bf(w4[j]);
  }
  if (tid < 64) bsm[tid] = bo[tid];

#pragma unroll
  for (int pass = 0; pass < 4; ++pass) {
    int i = pass * 2048 + tid * 8;
    int cch = i >> 7, col = i & 127;
    BF8 ov; ov.v = *(const v4u*)(obuf + (size_t)cch * kNV + p0 + col);
#pragma unroll
    for (int j = 0; j < 8; ++j) xsm[(col + j) * 72 + cch] = ov.s[j];
  }
  __syncthreads();

  const int wvid = tid >> 5;
  const int lane = tid & 31;
  const int nn   = wvid * 16 + (lane & 15);
  const int mhi  = (lane >> 4) << 3;

#pragma unroll
  for (int mb = 0; mb < 4; ++mb) {
    v8f acc = zero8();
#pragma unroll
    for (int kk = 0; kk < 64; kk += 32) {
      v16bf a = load_matA<72>(wsm, mb * 16, kk);
      v16bf b = load_matB<72>(xsm, kk, wvid * 16);
      acc = wmma_bf16(a, b, acc);
    }
#pragma unroll
    for (int r = 0; r < 8; ++r) {
      int co = mb * 16 + r + mhi;
      out[(size_t)co * kNV + p0 + nn] = acc[r] + bsm[co];
    }
  }
}

// ============================================================
extern "C" void kernel_launch(void* const* d_in, const int* in_sizes, int n_in,
                              void* d_out, int out_size, void* d_ws, size_t ws_size,
                              hipStream_t stream) {
  const float* x  = (const float*)d_in[0];
  const float* wq = (const float*)d_in[1];
  const float* bq = (const float*)d_in[2];
  const float* wk = (const float*)d_in[3];
  const float* bk = (const float*)d_in[4];
  const float* wv = (const float*)d_in[5];
  const float* bv = (const float*)d_in[6];
  const float* wo = (const float*)d_in[7];
  const float* bo = (const float*)d_in[8];

  // workspace: q | k | v as bf16 [C][D][W][H]  (3 * 96 MiB)
  unsigned short* qbuf = (unsigned short*)d_ws;
  unsigned short* kbuf = qbuf + (size_t)kC * kNV;
  unsigned short* vbuf = kbuf + (size_t)kC * kNV;

  qkv_proj_kernel<<<kNV / 128, 256, 0, stream>>>(x, wq, bq, wk, bk, wv, bv,
                                                 qbuf, kbuf, vbuf);

  // 4 bf16 tiles of 128x136 in dynamic LDS (~139 KB of the 320 KB/WGP)
  size_t lds2 = (size_t)4 * 128 * 136 * sizeof(unsigned short);
  attn_kernel<<<dim3(kD, kC), 256, lds2, stream>>>(qbuf, kbuf, vbuf, qbuf);

  out_proj_kernel<<<kNV / 128, 256, 0, stream>>>(qbuf, wo, bo, (float*)d_out);
}
